// CAPR_91199335563701
// MI455X (gfx1250) — matrix operand order repaired and checked by reading, not compiled, not run on hardware
//
#include <hip/hip_runtime.h>
#include <hip/hip_bf16.h>

typedef __attribute__((ext_vector_type(16))) __bf16          v16bf;
typedef __attribute__((ext_vector_type(8)))  float           v8f;
typedef __attribute__((ext_vector_type(4)))  float           f4v;
typedef __attribute__((ext_vector_type(8)))  unsigned int    u8v;

#define HIDDEN   4096
#define NEXP     64
#define TOPK     8
#define KC       128            // K-chunk staged in LDS per iteration
#define WAVES    4
#define TOK_WAVE 32             // two 16-row M-tiles per wave
#define TOK_WG   (TOK_WAVE * WAVES)   // 128 tokens per workgroup

// Pack two f32 into one VGPR of two bf16 (round-half-up: add 0x8000, take hi16).
// v_perm_b32 selector 0x07060302: out = { hi[31:16], lo[31:16] }.
__device__ __forceinline__ unsigned int pack_bf16(float lo, float hi) {
    unsigned int ulo = __builtin_bit_cast(unsigned int, lo) + 0x8000u;
    unsigned int uhi = __builtin_bit_cast(unsigned int, hi) + 0x8000u;
    return __builtin_amdgcn_perm(uhi, ulo, 0x07060302u);
}

// Build a 16x32 bf16 A-fragment (contiguous k order, same as the B fragments) from a f32 row.
// lane<16 holds K {0..15}; lane>=16 holds K {16..31} (k0 pre-offset).
__device__ __forceinline__ v16bf make_afrag(const float* __restrict__ row, int k0) {
    f4v a0 = *(const f4v*)(row + k0);
    f4v a1 = *(const f4v*)(row + k0 + 4);
    f4v a2 = *(const f4v*)(row + k0 + 8);
    f4v a3 = *(const f4v*)(row + k0 + 12);
    u8v au;
    au[0] = pack_bf16(a0.x, a0.y);
    au[1] = pack_bf16(a0.z, a0.w);
    au[2] = pack_bf16(a1.x, a1.y);
    au[3] = pack_bf16(a1.z, a1.w);
    au[4] = pack_bf16(a2.x, a2.y);
    au[5] = pack_bf16(a2.z, a2.w);
    au[6] = pack_bf16(a3.x, a3.y);
    au[7] = pack_bf16(a3.z, a3.w);
    return __builtin_bit_cast(v16bf, au);
}

__global__ __launch_bounds__(128) void moe_router_kernel(
    const float* __restrict__ x,       // [16384, 4096]
    const float* __restrict__ pk,      // [64, 4096]
    const float* __restrict__ gate,    // [64]
    float* __restrict__ out_w,         // [16384, 8]
    int*   __restrict__ out_e)         // [16384, 8]
{
    __shared__ __align__(32) unsigned short sB[NEXP][KC];   // proto_k chunk, bf16
    __shared__ float sLogit[TOK_WG][NEXP + 1];              // +1 pad vs bank conflicts

    const int tid  = threadIdx.x;
    const int wave = tid >> 5;
    const int lane = tid & 31;
    const int half = lane >> 4;        // 0: lanes 0-15, 1: lanes 16-31
    const int col  = lane & 15;        // N / M-row selector

    // staging assignment: thread -> (expert, 64-wide k slice)
    const int se = tid & 63;
    const int sk = (tid >> 6) * 64;

    const int tok0 = blockIdx.x * TOK_WG + wave * TOK_WAVE;
    const float* xrow0 = x + (size_t)(tok0 + col) * HIDDEN;
    const float* xrow1 = x + (size_t)(tok0 + 16 + col) * HIDDEN;

    v8f acc[2][4];
    #pragma unroll
    for (int m = 0; m < 2; ++m)
        #pragma unroll
        for (int n = 0; n < 4; ++n) acc[m][n] = (v8f)0.0f;

    for (int kc = 0; kc < HIDDEN; kc += KC) {
        __syncthreads();   // previous chunk fully consumed
        // ---- stage 64 x 128 proto_k chunk into LDS as bf16 ----
        {
            const float* src = pk + (size_t)se * HIDDEN + kc + sk;
            #pragma unroll
            for (int h = 0; h < 4; ++h) {           // four 16-float sub-slices
                u8v p;
                #pragma unroll
                for (int i = 0; i < 4; ++i) {
                    f4v v = *(const f4v*)(src + h * 16 + i * 4);
                    p[i*2+0] = pack_bf16(v.x, v.y);
                    p[i*2+1] = pack_bf16(v.z, v.w);
                }
                *(u8v*)&sB[se][sk + h * 16] = p;
            }
        }
        __syncthreads();

        // ---- 4 WMMA K-steps of 32 over the staged chunk ----
        #pragma unroll
        for (int ks = 0; ks < 4; ++ks) {
            // B fragments (shared by both M-tiles -> must stay live across 8 WMMAs)
            v16bf bv[4];
            #pragma unroll
            for (int n = 0; n < 4; ++n)
                bv[n] = *(const v16bf*)&sB[n * 16 + col][ks * 32 + (half ? 16 : 0)];

            const int k0 = kc + ks * 32 + (half ? 16 : 0);
            v16bf a0 = make_afrag(xrow0, k0);
            v16bf a1 = make_afrag(xrow1, k0);

            #pragma unroll
            for (int n = 0; n < 4; ++n)
                acc[0][n] = __builtin_amdgcn_wmma_f32_16x16x32_bf16(
                    false, a0, false, bv[n], (short)0, acc[0][n], false, false);
            #pragma unroll
            for (int n = 0; n < 4; ++n)
                acc[1][n] = __builtin_amdgcn_wmma_f32_16x16x32_bf16(
                    false, a1, false, bv[n], (short)0, acc[1][n], false, false);
        }
    }

    // ---- epilogue: scale, gate, relu -> LDS logits ----
    const float inv_sqrt_d = 0.015625f;   // 1/sqrt(4096)
    #pragma unroll
    for (int m = 0; m < 2; ++m) {
        #pragma unroll
        for (int n = 0; n < 4; ++n) {
            const float g = gate[n * 16 + col];
            #pragma unroll
            for (int j = 0; j < 8; ++j) {
                float v = acc[m][n][j] * inv_sqrt_d - g;
                v = v > 0.0f ? v : 0.0f;
                sLogit[wave * TOK_WAVE + m * 16 + half * 8 + j][n * 16 + col] = v;
            }
        }
    }
    __syncthreads();

    // ---- top-8 per token: one thread per token, 8 argmax passes ----
    {
        float* row = sLogit[tid];
        const size_t gt = (size_t)blockIdx.x * TOK_WG + tid;
        float* ow = out_w + gt * TOPK;
        int*   oe = out_e + gt * TOPK;
        #pragma unroll 1
        for (int i = 0; i < TOPK; ++i) {
            float bv = row[0];
            int   bi = 0;
            #pragma unroll 8
            for (int e = 1; e < NEXP; ++e) {
                float v = row[e];
                if (v > bv) { bv = v; bi = e; }
            }
            ow[i] = bv;
            oe[i] = bi;
            row[bi] = -1.0e30f;
        }
    }
}

extern "C" void kernel_launch(void* const* d_in, const int* in_sizes, int n_in,
                              void* d_out, int out_size, void* d_ws, size_t ws_size,
                              hipStream_t stream) {
    const float* x    = (const float*)d_in[0];   // [4,4096,4096]
    const float* pk   = (const float*)d_in[1];   // [64,4096]
    const float* gate = (const float*)d_in[2];   // [64]

    const int tokens = in_sizes[0] / HIDDEN;     // 16384
    float* out_w = (float*)d_out;                        // [tokens, 8] f32
    int*   out_e = (int*)((float*)d_out + (size_t)tokens * TOPK);  // [tokens, 8] i32

    const int blocks = tokens / TOK_WG;          // 128
    moe_router_kernel<<<blocks, 128, 0, stream>>>(x, pk, gate, out_w, out_e);
}
